// GraphSAGENet_38439957299734
// MI455X (gfx1250) — matrix operand order repaired and checked by reading, not compiled, not run on hardware
//
#include <hip/hip_runtime.h>
#include <hip/hip_bf16.h>

// ---------------------------------------------------------------------------
// GraphSAGE link predictor for MI455X (gfx1250, wave32, WMMA).
// All GEMM work goes through v_wmma_f32_16x16x32_f16 (fp32 accumulate).
// Fragment layouts follow CDNA5 ISA section 7.12.2 exactly:
//   A (16x32 f16): lane&15 = M row; lanes>=16 shift K by +8; vgprs 4..7 hold K+16.
//   B (32x16 f16): lane&15 = N col; same K order as A (lanes>=16 +8, vgprs 4..7 K+16).
//   C/D (16x16 f32): lane&15 = N; vgpr r -> M = r + 8*(lane>>4).
// GEMMs are register-blocked: one wave owns a 16-row strip and ALL output
// columns, so each A row is read from L2 exactly once per layer.
// ---------------------------------------------------------------------------

typedef _Float16 h16;
typedef __attribute__((ext_vector_type(16))) _Float16 v16h;
typedef __attribute__((ext_vector_type(8)))  float    v8f;

#define D_IN   64
#define D_H    128
#define D_OUTL 64   // layer-2 output dim

union FragA { v16h v; h16  e[16]; };
union FragC { v8f  v; float e[8]; };

// ------------------------------ utility kernels ----------------------------

__global__ void zero_f32(float* __restrict__ p, long n) {
  long i = (long)blockIdx.x * blockDim.x + threadIdx.x;
  if (i < n) p[i] = 0.0f;
}

__global__ void count_edges(const int* __restrict__ dst, float* __restrict__ cnt, int E) {
  int i = blockIdx.x * blockDim.x + threadIdx.x;
  if (i < E) atomicAdd(&cnt[dst[i]], 1.0f);
}

// scatter-add feat[src[e]] into agg[dst[e]] ; one thread per (edge, feature)
__global__ void scatter_feat(const float* __restrict__ feat, const int* __restrict__ src,
                             const int* __restrict__ dst, float* __restrict__ agg,
                             int E, int D) {
  long i = (long)blockIdx.x * blockDim.x + threadIdx.x;
  if (i >= (long)E * D) return;
  int e = (int)(i / D);
  int f = (int)(i - (long)e * D);
  atomicAdd(&agg[(long)dst[e] * D + f], feat[(long)src[e] * D + f]);
}

// convert W[K,N] (row-major f32) -> Wt[N,K] f16 (so B-fragment K-pairs are contiguous)
__global__ void cvt_w_t(const float* __restrict__ W, h16* __restrict__ Wt, int K, int N) {
  int i = blockIdx.x * blockDim.x + threadIdx.x;
  if (i >= K * N) return;
  int k = i / N;
  int n = i - k * N;
  Wt[(long)n * K + k] = (h16)W[i];
}

// ------------------------------ SAGE linear --------------------------------
// out[N, NT*16] = (agg/max(cnt,1)) @ Wl + x @ Wr + bias
// One wave per 16-row strip; NT accumulator tiles cover the full output width.
// grid = N/16, block = 32.
template <int K, int NT>
__global__ void sage_gemm_strip(const float* __restrict__ agg, const float* __restrict__ cnt,
                                const float* __restrict__ xr,
                                const h16* __restrict__ Wlt, const h16* __restrict__ Wrt,
                                const float* __restrict__ bias, float* __restrict__ out) {
  constexpr int Dout = NT * 16;
  int lane = threadIdx.x;
  int lo = lane & 15;
  int hi = lane >> 4;

  int rowA = blockIdx.x * 16 + lo;          // this lane's A-matrix row (M = lane&15)
  float rc = 1.0f / fmaxf(cnt[rowA], 1.0f);
  const float* aggRow = agg + (long)rowA * K;
  const float* xRow   = xr  + (long)rowA * K;

  int kbA = hi * 8;                          // A-frag K shift for upper half-wave
  int kbB = hi * 16;                         // unused; B frag uses kbA (same K order as A)

  FragC acc[NT];
#pragma unroll
  for (int nt = 0; nt < NT; ++nt) {
    float bn = bias[nt * 16 + lo];           // bias folded into accumulator init
#pragma unroll
    for (int r = 0; r < 8; ++r) acc[nt].e[r] = bn;
  }

#pragma unroll
  for (int kt = 0; kt < K; kt += 32) {
    FragA a, a2;
#pragma unroll
    for (int i = 0; i < 8; ++i) {
      int kk = kt + kbA + 2 * i + ((i >= 4) ? 8 : 0);   // A layout: vgprs 4..7 = K+16
      float2 va = *(const float2*)(aggRow + kk);
      float2 vx = *(const float2*)(xRow + kk);
      a.e[2*i]    = (h16)(va.x * rc);
      a.e[2*i+1]  = (h16)(va.y * rc);
      a2.e[2*i]   = (h16)vx.x;
      a2.e[2*i+1] = (h16)vx.y;
    }
#pragma unroll
    for (int nt = 0; nt < NT; ++nt) {
      int n = nt * 16 + lo;
      const h16* wl = Wlt + (long)n * K;
      const h16* wr = Wrt + (long)n * K;
      FragA b, b2;
#pragma unroll
      for (int i = 0; i < 8; ++i) {
        int kB = kt + kbA + 2 * i + ((i >= 4) ? 8 : 0); // B layout: same K order as A
        b.e[2*i]    = wl[kB];
        b.e[2*i+1]  = wl[kB + 1];
        b2.e[2*i]   = wr[kB];
        b2.e[2*i+1] = wr[kB + 1];
      }
      acc[nt].v = __builtin_amdgcn_wmma_f32_16x16x32_f16(false, a.v,  false, b.v,
                                                         (short)0, acc[nt].v, false, false);
      acc[nt].v = __builtin_amdgcn_wmma_f32_16x16x32_f16(false, a2.v, false, b2.v,
                                                         (short)0, acc[nt].v, false, false);
    }
  }

#pragma unroll
  for (int nt = 0; nt < NT; ++nt) {
    int n = nt * 16 + lo;
#pragma unroll
    for (int r = 0; r < 8; ++r) {
      int m = r + 8 * hi;                               // C/D layout
      out[(long)(blockIdx.x * 16 + m) * Dout + n] = acc[nt].e[r];
    }
  }
}

// ------------------------- fused edge MLP ----------------------------------
// Per wave: 16 edges.  A = h2[src] * h2[dst] (built on the fly),
// hidden = relu(A @ W1 + b1) kept entirely in registers (8 n-tiles of C),
// logit = hidden . W2 + b2 via per-lane dot + half-wave shfl_xor reduction.
__global__ void edge_mlp(const float* __restrict__ h2,
                         const int* __restrict__ src, const int* __restrict__ dst,
                         const h16* __restrict__ W1t, const float* __restrict__ b1,
                         const float* __restrict__ W2, const float* __restrict__ b2,
                         float* __restrict__ out) {
  int lane = threadIdx.x;
  int lo = lane & 15;
  int hi = lane >> 4;
  long eBase = (long)blockIdx.x * 16;
  long eIdx  = eBase + lo;

  int si = src[eIdx];
  int di = dst[eIdx];
  const float* srow = h2 + (long)si * D_OUTL;
  const float* drow = h2 + (long)di * D_OUTL;

  int kbA = hi * 8;
  int kbB = hi * 16;

  FragC acc[8];
#pragma unroll
  for (int nt = 0; nt < 8; ++nt)
#pragma unroll
    for (int r = 0; r < 8; ++r) acc[nt].e[r] = 0.0f;

#pragma unroll
  for (int kt = 0; kt < D_OUTL; kt += 32) {
    FragA a;
#pragma unroll
    for (int i = 0; i < 8; ++i) {
      int kk = kt + kbA + 2 * i + ((i >= 4) ? 8 : 0);
      float2 s = *(const float2*)(srow + kk);
      float2 d = *(const float2*)(drow + kk);
      a.e[2*i]   = (h16)(s.x * d.x);
      a.e[2*i+1] = (h16)(s.y * d.y);
    }
#pragma unroll
    for (int nt = 0; nt < 8; ++nt) {
      int n = nt * 16 + lo;
      const h16* w = W1t + (long)n * D_OUTL;
      FragA b;
#pragma unroll
      for (int i = 0; i < 8; ++i) {
        int kB = kt + kbA + 2 * i + ((i >= 4) ? 8 : 0);
        b.e[2*i]   = w[kB];
        b.e[2*i+1] = w[kB + 1];
      }
      acc[nt].v = __builtin_amdgcn_wmma_f32_16x16x32_f16(false, a.v, false, b.v,
                                                         (short)0, acc[nt].v, false, false);
    }
  }

  // bias + relu + dot with W2, per lane; row m = r + 8*hi, col n = nt*16 + lo
  float accr[8];
#pragma unroll
  for (int r = 0; r < 8; ++r) accr[r] = 0.0f;
#pragma unroll
  for (int nt = 0; nt < 8; ++nt) {
    int n = nt * 16 + lo;
    float bb = b1[n];
    float w2 = W2[n];
#pragma unroll
    for (int r = 0; r < 8; ++r) {
      float v = fmaxf(acc[nt].e[r] + bb, 0.0f);
      accr[r] += v * w2;
    }
  }
  // reduce over the 16 lanes of each half-wave (xor 1,2,4,8 keeps halves separate)
#pragma unroll
  for (int r = 0; r < 8; ++r) {
    accr[r] += __shfl_xor(accr[r], 1, 32);
    accr[r] += __shfl_xor(accr[r], 2, 32);
    accr[r] += __shfl_xor(accr[r], 4, 32);
    accr[r] += __shfl_xor(accr[r], 8, 32);
  }
  if (lo == 0) {
    float bb2 = b2[0];
#pragma unroll
    for (int r = 0; r < 8; ++r)
      out[eBase + r + 8 * hi] = accr[r] + bb2;
  }
}

// ------------------------------- launcher ----------------------------------

extern "C" void kernel_launch(void* const* d_in, const int* in_sizes, int n_in,
                              void* d_out, int out_size, void* d_ws, size_t ws_size,
                              hipStream_t stream) {
  const float* x   = (const float*)d_in[0];
  const int*   ei  = (const int*)d_in[1];
  const float* Wl1 = (const float*)d_in[2];
  const float* bl1 = (const float*)d_in[3];
  const float* Wr1 = (const float*)d_in[4];
  const float* Wl2 = (const float*)d_in[5];
  const float* bl2 = (const float*)d_in[6];
  const float* Wr2 = (const float*)d_in[7];
  const float* W1  = (const float*)d_in[8];
  const float* b1  = (const float*)d_in[9];
  const float* W2  = (const float*)d_in[10];
  const float* b2  = (const float*)d_in[11];
  float* out = (float*)d_out;

  const int N = in_sizes[0] / D_IN;   // 100000 (divisible by 16)
  const int E = in_sizes[1] / 2;      // 1200000 (divisible by 16)
  const int* src = ei;
  const int* dst = ei + E;

  // workspace layout (floats): cnt[N] | r1[N*64] (agg1, later h2) | h1[N*128] | agg2[N*128] | f16 weights
  float* cnt  = (float*)d_ws;
  float* r1   = cnt + N;
  float* h1   = r1 + (long)N * D_IN;
  float* agg2 = h1 + (long)N * D_H;
  h16* Wl1t = (h16*)(agg2 + (long)N * D_H);     // [128,64]
  h16* Wr1t = Wl1t + D_H * D_IN;                // [128,64]
  h16* Wl2t = Wr1t + D_H * D_IN;                // [64,128]
  h16* Wr2t = Wl2t + D_OUTL * D_H;              // [64,128]
  h16* W1t  = Wr2t + D_OUTL * D_H;              // [128,64]

  // tiny weight conversions (f32 -> transposed f16)
  cvt_w_t<<<(D_IN * D_H + 255) / 256, 256, 0, stream>>>(Wl1, Wl1t, D_IN, D_H);
  cvt_w_t<<<(D_IN * D_H + 255) / 256, 256, 0, stream>>>(Wr1, Wr1t, D_IN, D_H);
  cvt_w_t<<<(D_H * D_OUTL + 255) / 256, 256, 0, stream>>>(Wl2, Wl2t, D_H, D_OUTL);
  cvt_w_t<<<(D_H * D_OUTL + 255) / 256, 256, 0, stream>>>(Wr2, Wr2t, D_H, D_OUTL);
  cvt_w_t<<<(D_OUTL * D_H + 255) / 256, 256, 0, stream>>>(W1, W1t, D_OUTL, D_H);

  // ---- layer 1: mean aggregate + linear ----
  long z1 = (long)N + (long)N * D_IN;           // cnt + agg1 are contiguous
  zero_f32<<<(unsigned)((z1 + 255) / 256), 256, 0, stream>>>(cnt, z1);
  count_edges<<<(E + 255) / 256, 256, 0, stream>>>(dst, cnt, E);
  {
    long nthr = (long)E * D_IN;
    scatter_feat<<<(unsigned)((nthr + 255) / 256), 256, 0, stream>>>(x, src, dst, r1, E, D_IN);
  }
  sage_gemm_strip<D_IN, D_H / 16><<<N / 16, 32, 0, stream>>>(r1, cnt, x, Wl1t, Wr1t, bl1, h1);

  // ---- layer 2 ----
  long z2 = (long)N * D_H;
  zero_f32<<<(unsigned)((z2 + 255) / 256), 256, 0, stream>>>(agg2, z2);
  {
    long nthr = (long)E * D_H;
    scatter_feat<<<(unsigned)((nthr + 255) / 256), 256, 0, stream>>>(h1, src, dst, agg2, E, D_H);
  }
  // h2 reuses r1 (agg1 is dead here)
  sage_gemm_strip<D_H, D_OUTL / 16><<<N / 16, 32, 0, stream>>>(agg2, cnt, h1, Wl2t, Wr2t, bl2, r1);

  // ---- fused edge MLP: e = h2[src]*h2[dst]; relu(e@W1+b1)@W2+b2 ----
  edge_mlp<<<E / 16, 32, 0, stream>>>(r1, src, dst, W1t, b1, W2, b2, out);
}
